// GATLayer_44418551775894
// MI455X (gfx1250) — matrix operand order repaired and checked by reading, not compiled, not run on hardware
//
#include <hip/hip_runtime.h>
#include <stdint.h>

// ---------------------------------------------------------------------------
// GAT layer for MI455X (gfx1250, wave32, WMMA + TDM).
// bf16 V_WMMA_F32_16X16X32_BF16 with f32 accumulation for both GEMMs;
// f32 scores/softmax/output. One batch element per 256-thread block.
// x tile staged into LDS by the Tensor Data Mover (TENSORcnt-tracked DMA),
// landing as f32 into the (still dead) h buffer, then converted to bf16.
// ---------------------------------------------------------------------------

typedef __attribute__((ext_vector_type(16))) __bf16 v16bf;
typedef __attribute__((ext_vector_type(8)))  __bf16 v8bf;
typedef __attribute__((ext_vector_type(8)))  float  v8f;
typedef __attribute__((ext_vector_type(4)))  float  v4f;
typedef __attribute__((ext_vector_type(4)))  unsigned int v4u;
typedef __attribute__((ext_vector_type(8)))  int v8i;
typedef __attribute__((ext_vector_type(4)))  int v4i;

#define BATCH 2048
#define MROW  64
#define CH    256
#define XS_S  264   // bf16 row stride for x tile (33*16B, bank-skewed)
#define HS_S  260   // f32 row stride for h tile (TDM pad: +4 dwords per 256)
#define AT_S  72    // bf16 row stride for attn tile (9*16B)

#define HAVE_TDM (__has_builtin(__builtin_amdgcn_tensor_load_to_lds) && \
                  __has_builtin(__builtin_amdgcn_s_wait_tensorcnt))

static __device__ __forceinline__ v16bf catbf(v8bf lo, v8bf hi) {
  v16bf r;
#pragma unroll
  for (int i = 0; i < 8; ++i) { r[i] = lo[i]; r[i + 8] = hi[i]; }
  return r;
}

// ---------------------------------------------------------------------------
// Prepass: pack W (C x C row-major fp32) into WMMA-B bf16 layout (contiguous k order, as A uses):
//   Wb[kt][nt][lane][i], lane l<16 -> n=nt*16+l, k=kt*32+i ; l>=16 -> k+=16.
// B element (k=c, n=o) = W[o][c].  128 KB total, stays hot in L2.
// ---------------------------------------------------------------------------
__global__ __launch_bounds__(256) void pack_w_kernel(const float* __restrict__ W,
                                                     __bf16* __restrict__ Wb) {
  int idx = blockIdx.x * 256 + threadIdx.x;       // 65536 elements
  int i   = idx & 15;
  int l   = (idx >> 4) & 31;
  int nt  = (idx >> 9) & 15;
  int kt  = idx >> 13;
  int n   = nt * 16 + (l & 15);
  int k   = kt * 32 + ((l >> 4) << 4) + i;
  Wb[idx] = (__bf16)W[n * CH + k];
}

// ---------------------------------------------------------------------------
// Main kernel: one batch element per block (256 threads = 8 waves).
// ---------------------------------------------------------------------------
__global__ __launch_bounds__(256) void gat_kernel(const float* __restrict__ x,
                                                  const __bf16* __restrict__ Wb,
                                                  const float* __restrict__ aw,
                                                  float* __restrict__ out) {
  __shared__ __align__(16) __bf16 xs[MROW * XS_S];          // x tile, bf16
  __shared__ __align__(16) float  hs[MROW * HS_S];          // TDM landing / h f32
  __shared__ __align__(16) __bf16 hpk[2 * 16 * 32 * 16];    // h repacked (B-layout)
  __shared__ __align__(16) __bf16 at[MROW * AT_S];          // softmax(attn), bf16
  __shared__ float sp_i[MROW][4];
  __shared__ float sp_j[MROW][4];
  __shared__ float s_i[MROW];
  __shared__ float s_j[MROW];

  const int t    = threadIdx.x;
  const int lane = t & 31;
  const int wave = t >> 5;
  const int b    = blockIdx.x;
  const float* xb = x + (size_t)b * MROW * CH;

#if HAVE_TDM
  // --- stage x[b] (64x256 f32, 64KB) into hs via Tensor Data Mover ---
  // D# per CDNA5 ISA 8.3/8.4: 2D tile 256x64, data_size=4B,
  // pad 4 dwords after every 256 dwords -> LDS row stride 260 = HS_S.
  if (wave == 0) {
    uint64_t ga  = (uint64_t)(uintptr_t)xb;
    uint32_t lds = (uint32_t)(uintptr_t)(void*)hs;   // addr[31:0] = LDS offset
    v4u g0; v8i g1; v4i g2 = {}; v4i g3 = {}; v8i g4 = {};
    g0[0] = 1u;                                      // count=1 valid descriptor
    g0[1] = lds;                                     // lds_addr
    g0[2] = (uint32_t)ga;                            // global_addr[31:0]
    g0[3] = (uint32_t)((ga >> 32) & 0x01FFFFFFu)     // global_addr[56:32]
          | (2u << 30);                              // type=2 ("image")
    g1[0] = (int)((2u << 16)                         // data_size: 4 bytes
          | (1u << 20)                               // pad_enable
          | (7u << 22)                               // pad_interval: 256 dwords
          | (3u << 25));                             // pad_amount: 4 dwords
    g1[1] = (int)(256u << 16);                       // tensor_dim0 = 256
    g1[2] = (int)((uint32_t)MROW << 16);             // tensor_dim1 = 64
    g1[3] = (int)(256u << 16);                       // tile_dim0 = 256
    g1[4] = MROW;                                    // tile_dim1 = 64
    g1[5] = 256;                                     // tensor_dim0_stride = 256
    g1[6] = 0; g1[7] = 0;
    __builtin_amdgcn_tensor_load_to_lds(g0, g1, g2, g3, g4, 0);
    // overlap: pull this block's first packed-W lines toward the WGP
    __builtin_prefetch(Wb + (lane << 4), 0, 0);
    __builtin_amdgcn_s_wait_tensorcnt(0);
  }
  __syncthreads();
  // convert staged f32 rows (stride HS_S) -> bf16 xs (stride XS_S)
  for (int idx = t; idx < MROW * CH; idx += 256) {
    int m = idx >> 8, c = idx & 255;
    xs[m * XS_S + c] = (__bf16)hs[m * HS_S + c];
  }
  __syncthreads();
#else
  // --- fallback: coalesced float4 global loads + convert ---
  for (int i4 = t; i4 < MROW * CH / 4; i4 += 256) {
    int m = i4 >> 6;
    int c = (i4 & 63) << 2;
    v4f v = *(const v4f*)(xb + m * CH + c);
    __bf16* dst = &xs[m * XS_S + c];
    dst[0] = (__bf16)v.x; dst[1] = (__bf16)v.y;
    dst[2] = (__bf16)v.z; dst[3] = (__bf16)v.w;
  }
  __syncthreads();
#endif

  const int arow = lane & 15;      // A-matrix row within tile (M = lane%16)
  const int asel = lane >> 4;      // K-phase select per CDNA5 A layout

  // --- GEMM1: h = x @ W^T   (64x256x256, bf16 WMMA, f32 acc) ---
  for (int tile = wave; tile < 64; tile += 8) {
    const int mt = tile & 3;       // 4 M-tiles (constant per wave)
    const int nt = tile >> 2;      // 16 N-tiles
    v8f acc = {};
#pragma unroll
    for (int ks = 0; ks < 8; ++ks) {
      // A: lanes<16 hold K = base..base+15; lanes>=16 hold K = base+16..base+31.
      const __bf16* ap = &xs[(mt * 16 + arow) * XS_S + ks * 32 + asel * 16];
      v16bf a = catbf(*(const v8bf*)ap, *(const v8bf*)(ap + 8));
      // B: pre-packed, lane-contiguous 32B chunk (two b128 loads, coalesced).
      const __bf16* bp = Wb + ((((ks * 16 + nt) * 32) + lane) << 4);
      v16bf bm = catbf(*(const v8bf*)bp, *(const v8bf*)(bp + 8));
      acc = __builtin_amdgcn_wmma_f32_16x16x32_bf16(
          false, a, false, bm, (short)0, acc, false, false);
    }
    // D layout: VGPR r -> row r (lanes<16) / r+8 (lanes>=16), col = lane%16.
#pragma unroll
    for (int r = 0; r < 8; ++r) {
      int row = mt * 16 + ((lane < 16) ? r : (r + 8));
      hs[row * HS_S + nt * 16 + (lane & 15)] = acc[r];
    }
  }
  __syncthreads();

  // --- scores: partial dot products h[m,:]·a_i, h[m,:]·a_j (f32) ---
  {
    int m = t >> 2, seg = t & 3;
    float si = 0.f, sj = 0.f;
    const float* hrow = &hs[m * HS_S + seg * 64];
    const float* ai = aw + seg * 64;
    const float* aj = aw + CH + seg * 64;
#pragma unroll 8
    for (int c = 0; c < 64; ++c) {
      float hv = hrow[c];
      si += hv * ai[c];
      sj += hv * aj[c];
    }
    sp_i[m][seg] = si; sp_j[m][seg] = sj;
  }
  // --- repack h (f32 LDS) -> bf16 B layout (contiguous k) for GEMM2 ---
  for (int cid = t; cid < 1024; cid += 256) {
    int l  = cid & 31;
    int nt = (cid >> 5) & 15;
    int kt = cid >> 9;
    int n  = nt * 16 + (l & 15);
    int kb = kt * 32 + ((l >> 4) << 4);
    __bf16* dst = &hpk[cid << 4];
#pragma unroll
    for (int i = 0; i < 16; ++i) dst[i] = (__bf16)hs[(kb + i) * HS_S + n];
  }
  __syncthreads();

  if (t < MROW) {
    s_i[t] = sp_i[t][0] + sp_i[t][1] + sp_i[t][2] + sp_i[t][3];
    s_j[t] = sp_j[t][0] + sp_j[t][1] + sp_j[t][2] + sp_j[t][3];
  }
  __syncthreads();

  // --- leaky_relu + row softmax (one row per thread, f32) ---
  if (t < MROW) {
    const float vi = s_i[t];
    float mx = -3.0e38f;
    for (int j = 0; j < MROW; ++j) {
      float e = vi + s_j[j];
      e = (e > 0.f) ? e : 0.2f * e;
      mx = fmaxf(mx, e);
    }
    float sum = 0.f;
    __bf16* arow_p = &at[t * AT_S];
    for (int j = 0; j < MROW; ++j) {
      float e = vi + s_j[j];
      e = (e > 0.f) ? e : 0.2f * e;
      float ex = __expf(e - mx);
      sum += ex;
      arow_p[j] = (__bf16)ex;
    }
    float r = 1.f / sum;
    for (int j = 0; j < MROW; ++j)
      arow_p[j] = (__bf16)((float)arow_p[j] * r);
  }
  __syncthreads();

  // --- GEMM2: out = relu(attn @ h)   (64x256x64, bf16 WMMA, f32 acc) ---
  float* ob = out + (size_t)b * MROW * CH;
  for (int tile = wave; tile < 64; tile += 8) {
    const int mt = tile & 3;
    const int nt = tile >> 2;
    v8f acc = {};
#pragma unroll
    for (int ks = 0; ks < 2; ++ks) {
      const __bf16* ap = &at[(mt * 16 + arow) * AT_S + ks * 32 + asel * 16];
      v16bf a = catbf(*(const v8bf*)ap, *(const v8bf*)(ap + 8));
      const __bf16* bp = &hpk[((((ks * 16 + nt) * 32) + lane) << 4)];
      v16bf bm = catbf(*(const v8bf*)bp, *(const v8bf*)(bp + 8));
      acc = __builtin_amdgcn_wmma_f32_16x16x32_bf16(
          false, a, false, bm, (short)0, acc, false, false);
    }
#pragma unroll
    for (int r = 0; r < 8; ++r) {
      int row = mt * 16 + ((lane < 16) ? r : (r + 8));
      float v = acc[r];
      ob[row * CH + nt * 16 + (lane & 15)] = (v > 0.f) ? v : 0.f;
    }
  }
}

// ---------------------------------------------------------------------------
extern "C" void kernel_launch(void* const* d_in, const int* in_sizes, int n_in,
                              void* d_out, int out_size, void* d_ws, size_t ws_size,
                              hipStream_t stream) {
  const float* x  = (const float*)d_in[0];   // node_features (B,M,C) f32
  const float* W  = (const float*)d_in[1];   // W (C,C) f32
  const float* aw = (const float*)d_in[2];   // attn_w (2C,) f32
  __bf16* Wb = (__bf16*)d_ws;                // 128 KB packed bf16 W

  pack_w_kernel<<<256, 256, 0, stream>>>(W, Wb);
  gat_kernel<<<BATCH, 256, 0, stream>>>(x, Wb, aw, (float*)d_out);
}
